// MACE_42210938585319
// MI455X (gfx1250) — matrix operand order repaired and checked by reading, not compiled, not run on hardware
//
#include <hip/hip_runtime.h>
#include <math.h>

// ================= problem constants =================
constexpr int NN  = 5000;     // nodes
constexpr int NE  = 120000;   // edges
constexpr int NG  = 16;       // graphs
constexpr int NEL = 10;       // elements
constexpr int CH  = 64;       // channels
constexpr float R_MAX = 5.0f;
constexpr float AVGN  = 24.0f;

typedef __attribute__((ext_vector_type(16))) _Float16 v16h;
typedef __attribute__((ext_vector_type(8)))  float    v8f;

// ================= workspace layout (float elements) =================
constexpr long OFF_CG  = 0;                         // 64 keys * 343 floats
constexpr long OFF_Y   = 22016;                     // NE*16 spherical harmonics
constexpr long OFF_EF  = OFF_Y  + (long)NE*16;      // NE*8 radial features
constexpr long OFF_FA0 = OFF_EF + (long)NE*8;       // layer0 input f0 (NN*64)
constexpr long OFF_FB0 = OFF_FA0 + (long)NN*CH;     // layer0 out / layer1 in, l=0
constexpr long OFF_FB1 = OFF_FB0 + (long)NN*CH;     // layer0 out / layer1 in, l=1 (NN*64*3)
constexpr long OFF_FC0 = OFF_FB1 + (long)NN*CH*3;   // layer1 out l=0
constexpr long OFF_FU0 = OFF_FC0 + (long)NN*CH;     // up-projected f, l=0
constexpr long OFF_FU1 = OFF_FU0 + (long)NN*CH;     // up-projected f, l=1
constexpr long OFF_MSG = OFF_FU1 + (long)NN*CH*3;   // messages, all lo
constexpr long MSGSZ   = (long)NN*CH*(1+3+5+7);     // 5,120,000
constexpr long OFF_MID = OFF_MSG + MSGSZ;
constexpr long OFF_AC0 = OFF_MID + MSGSZ;
constexpr long OFF_AC1 = OFF_AC0 + (long)NN*CH;
constexpr long OFF_SC  = OFF_AC1 + (long)NN*CH*3;

// sub-offsets inside msg/mid blocks, per lo
constexpr long MO0 = 0;
constexpr long MO1 = (long)NN*CH;
constexpr long MO2 = MO1 + (long)NN*CH*3;
constexpr long MO3 = MO2 + (long)NN*CH*5;

__device__ __forceinline__ int cg_key(int l1,int l2,int l3){ return (l1*4+l2)*4+l3; }
__device__ __forceinline__ float siluf(float x){ return x / (1.f + __expf(-x)); }

__device__ __forceinline__ v8f wmma_f16(v16h a, v16h b, v8f c){
  // D = A(16x32 f16) * B(32x16 f16) + C(f32)
  return __builtin_amdgcn_wmma_f32_16x16x32_f16(false, a, false, b, (short)0, c, false, false);
}

// ================= utility =================
__global__ void k_zero(float* __restrict__ p, long n){
  long i = (long)blockIdx.x*blockDim.x + threadIdx.x;
  long st = (long)gridDim.x*blockDim.x;
  for(; i<n; i+=st) p[i]=0.f;
}

// ================= CG tables (exact, computed on device in fp64) =================
__device__ double dfact(int n){ double r=1.0; for(int i=2;i<=n;++i) r*=i; return r; }

__device__ double su2cg(int j1,int m1,int j2,int m2,int j3,int m3){
  if(m3 != m1+m2) return 0.0;
  int vmin = max(max(-j1+j2+m3, -j1+m1), 0);
  int vmax = min(min(j2+j3+m1, j3-j1+j2), j3+m3);
  double c = sqrt((2.0*j3+1.0)*dfact(j3+j1-j2)*dfact(j3-j1+j2)*dfact(j1+j2-j3)
                  *dfact(j3+m3)*dfact(j3-m3)
                  /(dfact(j1+j2+j3+1)*dfact(j1-m1)*dfact(j1+m1)*dfact(j2-m2)*dfact(j2+m2)));
  double s=0.0;
  for(int v=vmin; v<=vmax; ++v){
    double sg = ((v+j2+m2)&1)? -1.0 : 1.0;
    s += sg * dfact(j2+j3+m1-v)*dfact(j1-m1+v)
        /(dfact(v)*dfact(j3-j1+j2-v)*dfact(j3+m3-v)*dfact(v+j1-j2-m3));
  }
  return c*s;
}

__device__ void build_q(int l, double qr[7][7], double qi[7][7]){
  for(int i=0;i<7;++i) for(int j=0;j<7;++j){ qr[i][j]=0.0; qi[i][j]=0.0; }
  const double inv = 0.7071067811865475244;
  for(int m=-l; m<0; ++m){ qr[l+m][l-m] = inv; qi[l+m][l+m] = -inv; }
  qr[l][l] = 1.0;
  for(int m=1; m<=l; ++m){
    double s = (m&1)? -1.0 : 1.0;
    qr[l+m][l+m] = s*inv; qi[l+m][l-m] = s*inv;
  }
  int ph = l & 3;  // multiply by (-1j)^l
  for(int i=0;i<7;++i) for(int j=0;j<7;++j){
    double r=qr[i][j], m=qi[i][j];
    if(ph==1){ qr[i][j]= m; qi[i][j]=-r; }
    else if(ph==2){ qr[i][j]=-r; qi[i][j]=-m; }
    else if(ph==3){ qr[i][j]=-m; qi[i][j]= r; }
  }
}

__global__ void k_cg_init(float* __restrict__ cg){
  int t = threadIdx.x;
  if(t>=64) return;
  int l1=(t>>4)&3, l2=(t>>2)&3, l3=t&3;
  for(int i=0;i<343;++i) cg[(long)t*343+i]=0.f;
  int lmin = (l1>l2)? l1-l2 : l2-l1;
  int lmax = l1+l2; if(lmax>3) lmax=3;
  if(l3<lmin || l3>lmax) return;
  int n1=2*l1+1, n2=2*l2+1, n3=2*l3+1;
  double cc[7][7][7];
  for(int a=0;a<n1;++a) for(int b=0;b<n2;++b) for(int c=0;c<n3;++c)
    cc[a][b][c] = su2cg(l1,a-l1,l2,b-l2,l3,c-l3);
  double q1r[7][7],q1i[7][7],q2r[7][7],q2i[7][7],q3r[7][7],q3i[7][7];
  build_q(l1,q1r,q1i); build_q(l2,q2r,q2i); build_q(l3,q3r,q3i);
  for(int i=0;i<n1;++i) for(int j=0;j<n2;++j) for(int kk=0;kk<n3;++kk){
    double sr=0.0;
    for(int a=0;a<n1;++a) for(int b=0;b<n2;++b){
      double ar=q1r[a][i], ai=q1i[a][i];
      double br=q2r[b][j], bi=q2i[b][j];
      double pr=ar*br-ai*bi, pi=ar*bi+ai*br;
      for(int c=0;c<n3;++c){
        double v = cc[a][b][c]; if(v==0.0) continue;
        double crr=q3r[c][kk], cii=-q3i[c][kk];   // conj
        sr += v*(pr*crr - pi*cii);
      }
    }
    cg[(long)t*343 + (i*n2+j)*n3 + kk] = (float)sr;
  }
}

// ================= edge geometry: Y (l<=3) and Bessel*cutoff =================
__global__ void k_geom(const float* __restrict__ pos, const int* __restrict__ eidx,
                       float* __restrict__ Yb, float* __restrict__ EFb){
  int e = blockIdx.x*blockDim.x + threadIdx.x;
  if(e>=NE) return;
  int s = eidx[e], rcv = eidx[NE+e];
  float vx=pos[rcv*3+0]-pos[s*3+0];
  float vy=pos[rcv*3+1]-pos[s*3+1];
  float vz=pos[rcv*3+2]-pos[s*3+2];
  float r = sqrtf(vx*vx+vy*vy+vz*vz)+1e-9f;
  float x=vx/r, y=vy/r, z=vz/r;
  const float s3=1.7320508075688772f, s5=2.23606797749979f, s15=3.872983346207417f;
  const float c33=2.091650066335189f;   // sqrt(70)/4
  const float c32=10.246950765959598f;  // sqrt(105)
  const float c31=1.6201851746019651f;  // sqrt(42)/4
  const float c30=1.3228756555322954f;  // sqrt(7)/2
  float* Yr = Yb + (long)e*16;
  Yr[0]=1.f;
  Yr[1]=s3*y; Yr[2]=s3*z; Yr[3]=s3*x;
  Yr[4]=s15*x*y; Yr[5]=s15*y*z; Yr[6]=0.5f*s5*(3.f*z*z-1.f);
  Yr[7]=s15*x*z; Yr[8]=0.5f*s15*(x*x-y*y);
  Yr[9]=c33*y*(3.f*x*x-y*y); Yr[10]=c32*x*y*z; Yr[11]=c31*y*(5.f*z*z-1.f);
  Yr[12]=c30*(5.f*z*z*z-3.f*z); Yr[13]=c31*x*(5.f*z*z-1.f);
  Yr[14]=0.5f*c32*z*(x*x-y*y); Yr[15]=c33*x*(x*x-3.f*y*y);
  float xx = r/R_MAX;
  float x5=xx*xx*xx*xx*xx;
  float cut = (xx<1.f)? (1.f - 21.f*x5 + 35.f*x5*xx - 15.f*x5*xx*xx) : 0.f;
  float pref = sqrtf(2.f/R_MAX)/r * cut;
  float* EFr = EFb + (long)e*8;
  const float PIov = 3.14159265358979323846f / R_MAX;
  #pragma unroll
  for(int n=1;n<=8;++n) EFr[n-1] = pref * sinf((float)n * PIov * r);
}

// ================= node embedding + E0 energy =================
__global__ void k_embed(const float* __restrict__ na, const float* __restrict__ Wemb,
                        const float* __restrict__ E0v, const int* __restrict__ batch,
                        float* __restrict__ f0, float* __restrict__ out){
  int t = blockIdx.x*blockDim.x + threadIdx.x;
  if(t>=NN*CH) return;
  int n=t>>6, k=t&63;
  int spec=0; float best=-1e30f;
  for(int e=0;e<NEL;++e){ float v=na[(long)n*NEL+e]; if(v>best){best=v;spec=e;} }
  f0[t] = Wemb[(long)spec*CH+k];
  if(k==0) atomicAdd(&out[batch[n]], E0v[spec]);
}

// ================= generic node-level GEMM via WMMA =================
// out[n,k,m] = sum_c src[n,c,m]*scale * W[c,k]   (+ addv[n,k] when M==1)
// rows = (n,m) pairs; block = 4 waves x (16 rows x 16 cols).
// M and HAS_ADD are compile-time. Full-tile blocks take a block-uniform
// unguarded path (no per-element exec-mask toggles); only the single tail
// block uses clamped loads / guarded stores.
template<int M, bool HAS_ADD>
__global__ __launch_bounds__(128) void k_gemm_nodes(
    const float* __restrict__ src, const float* __restrict__ W,
    float* __restrict__ dst, const float* __restrict__ addv,
    int nRows, float scale)
{
  __shared__ _Float16 sS[16][64];    // A tile: (row-in-tile, c)
  __shared__ _Float16 sWtT[64][64];  // weights transposed: [k][c]
  int tid=threadIdx.x, lane=tid&31, wave=tid>>5;
  int r15=lane&15, hi=lane>>4, cb=wave*16;
  int kb8=hi*8, kb16=hi*16;
  int rowBase = blockIdx.x*16;
  bool full = (rowBase + 16 <= nRows);   // block-uniform

  // stage weights transposed (f32 -> f16): sWtT[col][c] = W[c*64+col]
  #pragma unroll
  for(int i=0;i<32;++i){
    int idx = tid + i*128;
    int c = idx>>6, col = idx&63;
    sWtT[col][c] = (_Float16)W[idx];
  }
  // stage A tile
  if(full){
    #pragma unroll
    for(int i=0;i<8;++i){
      int idx = tid + i*128;          // 0..1023
      int rT = idx>>6, c = idx&63;
      int row = rowBase + rT;
      int n = row / M, m = row % M;   // constexpr M -> cheap
      sS[rT][c] = (_Float16)(src[((long)n*CH + c)*M + m]*scale);
    }
  } else {
    #pragma unroll
    for(int i=0;i<8;++i){
      int idx = tid + i*128;
      int rT = idx>>6, c = idx&63;
      int row = rowBase + rT;
      int ok  = (row < nRows);
      int rc  = ok ? row : (nRows-1); // clamped, always in-bounds
      int n = rc / M, m = rc % M;
      float v = src[((long)n*CH + c)*M + m];
      sS[rT][c] = (_Float16)(ok ? v*scale : 0.f);
    }
  }
  __syncthreads();

  v8f acc = {0.f,0.f,0.f,0.f,0.f,0.f,0.f,0.f};
  #pragma unroll
  for(int kk=0;kk<64;kk+=32){
    v16h a,b;
    #pragma unroll
    for(int j=0;j<16;++j){ int k = kk + kb16 + j; a[j]=sS[r15][k]; }
    #pragma unroll
    for(int j=0;j<16;++j)  b[j]=sWtT[cb+r15][kk+kb16+j];
    acc = wmma_f16(a,b,acc);
  }

  int col = cb + r15;
  if(full){
    #pragma unroll
    for(int v=0;v<8;++v){
      int gr = rowBase + v + 8*hi;
      int nn2 = gr / M, mm2 = gr % M;
      float val = acc[v];
      if constexpr(HAS_ADD) val += addv[(long)nn2*CH + col];
      dst[((long)nn2*CH + col)*M + mm2] = val;
    }
  } else {
    #pragma unroll
    for(int v=0;v<8;++v){
      int gr = rowBase + v + 8*hi;
      if(gr < nRows){
        int nn2 = gr / M, mm2 = gr % M;
        float val = acc[v];
        if constexpr(HAS_ADD) val += addv[(long)nn2*CH + col];
        dst[((long)nn2*CH + col)*M + mm2] = val;
      }
    }
  }
}

// ================= fused edge kernel: radial MLP (WMMA) + messages =================
// one WMMA step over K=64, weights pre-staged transposed in LDS
__device__ __forceinline__ void mlp64_step(const _Float16 (*in)[64],
    const _Float16 (*wt)[64], _Float16 (*outp)[64],
    int r15,int hi,int cb,int kb8,int kb16)
{
  v16h a,b; v8f acc = {0.f,0.f,0.f,0.f,0.f,0.f,0.f,0.f};
  #pragma unroll
  for(int kk=0;kk<64;kk+=32){
    #pragma unroll
    for(int j=0;j<16;++j){ int k=kk+kb16+j; a[j]=in[r15][k]; }
    #pragma unroll
    for(int j=0;j<16;++j) b[j]=wt[cb+r15][kk+kb16+j];
    acc = wmma_f16(a,b,acc);
  }
  #pragma unroll
  for(int v=0;v<8;++v){ int row=v+8*hi; outp[row][cb+r15]=(_Float16)siluf(acc[v]); }
}

constexpr int PT0[4][3]  = {{0,0,0},{0,1,1},{0,2,2},{0,3,3}};
constexpr int PT1[10][3] = {{0,0,0},{0,1,1},{0,2,2},{0,3,3},{1,0,1},
                            {1,1,0},{1,1,2},{1,2,1},{1,2,3},{1,3,2}};

template<int LAYER>
__global__ __launch_bounds__(128) void k_edge(
    const float* __restrict__ cg, const float* __restrict__ Yb, const float* __restrict__ EFb,
    const float* __restrict__ fu0, const float* __restrict__ fu1,
    const float* __restrict__ rad0, const float* __restrict__ rad1,
    const float* __restrict__ rad2, const float* __restrict__ rad3,
    const int* __restrict__ eidx, float* __restrict__ msg)
{
  constexpr int NPATH = (LAYER==0)?4:10;
  constexpr int P = NPATH*CH;
  __shared__ _Float16 sA[16][64];
  __shared__ _Float16 sB[16][64];
  __shared__ _Float16 sRt[64][64];   // current-step weights, transposed [k][c]
  __shared__ float sW[16][P];
  int tid=threadIdx.x, lane=tid&31, wave=tid>>5;
  int r15=lane&15, hi=lane>>4, cb=wave*16;
  int kb8=hi*8, kb16=hi*16;
  int ebase=blockIdx.x*16;

  // stage rad0 transposed with K padded 8->64 by zeros: sRt[col][c]
  #pragma unroll
  for(int i=0;i<32;++i){
    int idx = tid + i*128;
    int c = idx>>6, col = idx&63;
    sRt[col][c] = (c<8)? (_Float16)rad0[(long)c*CH+col] : (_Float16)0.f;
  }
  // stage ef tile (16 edges x 8) into sA, zero-pad K 8..31
  #pragma unroll
  for(int i=0;i<4;++i){
    int idx = tid + i*128;          // 0..511 -> 16 x 32
    int e = idx>>5, c = idx&31;
    sA[e][c] = (c<8)? (_Float16)EFb[(long)(ebase+e)*8 + c] : (_Float16)0.f;
  }
  __syncthreads();

  // MLP step 0: (16x8 padded to 16x32) x (32x64)
  {
    v16h a,b; v8f acc = {0.f,0.f,0.f,0.f,0.f,0.f,0.f,0.f};
    #pragma unroll
    for(int j=0;j<16;++j){ int k=kb16+j; a[j]=sA[r15][k]; }
    #pragma unroll
    for(int j=0;j<16;++j)  b[j]=sRt[cb+r15][kb16+j];
    acc = wmma_f16(a,b,acc);
    #pragma unroll
    for(int v=0;v<8;++v){ int row=v+8*hi; sB[row][cb+r15]=(_Float16)siluf(acc[v]); }
  }
  __syncthreads();
  // stage rad1 transposed
  #pragma unroll
  for(int i=0;i<32;++i){
    int idx = tid + i*128; int c = idx>>6, col = idx&63;
    sRt[col][c] = (_Float16)rad1[idx];
  }
  __syncthreads();
  mlp64_step(sB, sRt, sA, r15,hi,cb,kb8,kb16);   // step 1: sB -> sA
  __syncthreads();
  // stage rad2 transposed
  #pragma unroll
  for(int i=0;i<32;++i){
    int idx = tid + i*128; int c = idx>>6, col = idx&63;
    sRt[col][c] = (_Float16)rad2[idx];
  }
  __syncthreads();
  mlp64_step(sA, sRt, sB, r15,hi,cb,kb8,kb16);   // step 2: sA -> sB
  __syncthreads();

  // MLP step 3: (16x64) x (64xP), no activation, fp32 result in LDS.
  // rad3 is grid-shared (L2-resident) -> per-lane global reads.
  for(int ct=wave; ct<P/16; ct+=4){
    v16h a,b; v8f acc = {0.f,0.f,0.f,0.f,0.f,0.f,0.f,0.f};
    #pragma unroll
    for(int kk=0;kk<64;kk+=32){
      #pragma unroll
      for(int j=0;j<16;++j){ int k=kk+kb16+j; a[j]=sB[r15][k]; }
      #pragma unroll
      for(int j=0;j<16;++j) b[j]=(_Float16)rad3[(long)(kk+kb16+j)*P + ct*16 + r15];
      acc = wmma_f16(a,b,acc);
    }
    #pragma unroll
    for(int v=0;v<8;++v) sW[v+8*hi][ct*16+r15]=acc[v];
  }
  __syncthreads();

  // ---- message phase: 8 threads per edge, 8 channels each ----
  int eloc = tid>>3;
  long e = ebase + eloc;
  int snd = eidx[e], rcv = eidx[NE + e];
  float Ysh[16];
  #pragma unroll
  for(int i=0;i<16;++i) Ysh[i]=Yb[e*16+i];
  int k0 = (tid&7)*8;
  float g0[8];
  #pragma unroll
  for(int q=0;q<8;++q) g0[q]=fu0[(long)snd*CH + k0+q];
  float g1[8][3] = {};
  if constexpr(LAYER==1){
    #pragma unroll
    for(int q=0;q<8;++q)
      #pragma unroll
      for(int aa=0;aa<3;++aa) g1[q][aa]=fu1[((long)snd*CH + (k0+q))*3 + aa];
  }
  for(int q=0;q<8;++q){
    int k=k0+q;
    float msum[16];
    #pragma unroll
    for(int i=0;i<16;++i) msum[i]=0.f;
    #pragma unroll
    for(int p=0;p<NPATH;++p){
      int l1 = (LAYER==0)? PT0[p][0] : PT1[p][0];
      int l2 = (LAYER==0)? PT0[p][1] : PT1[p][1];
      int lo = (LAYER==0)? PT0[p][2] : PT1[p][2];
      float wv = sW[eloc][p*CH + k];
      const float* g = cg + (long)cg_key(l1,l2,lo)*343;
      int n1=2*l1+1, n2=2*l2+1, n3=2*lo+1;
      int yo=l2*l2, oo=lo*lo;
      for(int aa=0; aa<n1; ++aa){
        float ga = (l1==0)? g0[q] : g1[q][aa];
        for(int bb=0; bb<n2; ++bb){
          float gy = ga * Ysh[yo+bb] * wv;
          for(int c=0;c<n3;++c) msum[oo+c] += gy * g[(aa*n2+bb)*n3+c];
        }
      }
    }
    long nb = (long)rcv*CH + k;
    atomicAdd(&msg[MO0 + nb], msum[0]);
    #pragma unroll
    for(int c=0;c<3;++c) atomicAdd(&msg[MO1 + nb*3 + c], msum[1+c]);
    #pragma unroll
    for(int c=0;c<5;++c) atomicAdd(&msg[MO2 + nb*5 + c], msum[4+c]);
    #pragma unroll
    for(int c=0;c<7;++c) atomicAdd(&msg[MO3 + nb*7 + c], msum[9+c]);
  }
}

// ================= per-node: skip (species-gated) + CG tensor products =================
__global__ void k_node_products(int layer,
    const float* __restrict__ na, const float* __restrict__ f0,
    const float* __restrict__ skipW, const float* __restrict__ mid,
    const float* __restrict__ pw1_0, const float* __restrict__ pw1_1,
    const float* __restrict__ pw2_00, const float* __restrict__ pw2_11,
    const float* __restrict__ pw2_22, const float* __restrict__ pw2_33,
    const float* __restrict__ pw2_01, const float* __restrict__ pw2_12,
    const float* __restrict__ pw2_23,
    const float* __restrict__ cg,
    float* __restrict__ acc0, float* __restrict__ acc1, float* __restrict__ sc)
{
  int t = blockIdx.x*blockDim.x + threadIdx.x;
  if(t >= NN*CH) return;
  int n=t>>6, k=t&63;
  int spec=0; float best=-1e30f;
  for(int e=0;e<NEL;++e){ float v=na[(long)n*NEL+e]; if(v>best){best=v;spec=e;} }
  long sk = (long)spec*CH + k;
  float s=0.f;
  for(int c=0;c<CH;++c) s += f0[(long)n*CH+c]*skipW[((long)spec*CH+c)*CH+k];
  sc[t]=s;
  float m0 = mid[MO0 + t];
  float m1[3],m2[5],m3[7];
  for(int a=0;a<3;++a) m1[a]=mid[MO1 + (long)t*3+a];
  for(int a=0;a<5;++a) m2[a]=mid[MO2 + (long)t*5+a];
  for(int a=0;a<7;++a) m3[a]=mid[MO3 + (long)t*7+a];

  float a0 = m0*pw1_0[sk];
  { const float* g=cg+(long)cg_key(0,0,0)*343; a0 += m0*m0*g[0]*pw2_00[sk]; }
  { const float* g=cg+(long)cg_key(1,1,0)*343; float tt=0;
    for(int a=0;a<3;++a) for(int b=0;b<3;++b) tt+=m1[a]*m1[b]*g[a*3+b];
    a0 += tt*pw2_11[sk]; }
  { const float* g=cg+(long)cg_key(2,2,0)*343; float tt=0;
    for(int a=0;a<5;++a) for(int b=0;b<5;++b) tt+=m2[a]*m2[b]*g[a*5+b];
    a0 += tt*pw2_22[sk]; }
  { const float* g=cg+(long)cg_key(3,3,0)*343; float tt=0;
    for(int a=0;a<7;++a) for(int b=0;b<7;++b) tt+=m3[a]*m3[b]*g[a*7+b];
    a0 += tt*pw2_33[sk]; }
  acc0[t]=a0;

  if(layer==0){
    float w1v=pw1_1[sk];
    float a1[3]; for(int c=0;c<3;++c) a1[c]=m1[c]*w1v;
    { const float* g=cg+(long)cg_key(0,1,1)*343; float w=pw2_01[sk];
      for(int b=0;b<3;++b){ float p=m0*m1[b]*w;
        for(int c=0;c<3;++c) a1[c]+= p*g[b*3+c]; } }
    { const float* g=cg+(long)cg_key(1,2,1)*343; float w=pw2_12[sk];
      for(int a=0;a<3;++a) for(int b=0;b<5;++b){ float p=m1[a]*m2[b]*w;
        for(int c=0;c<3;++c) a1[c]+= p*g[(a*5+b)*3+c]; } }
    { const float* g=cg+(long)cg_key(2,3,1)*343; float w=pw2_23[sk];
      for(int a=0;a<5;++a) for(int b=0;b<7;++b){ float p=m2[a]*m3[b]*w;
        for(int c=0;c<3;++c) a1[c]+= p*g[(a*7+b)*3+c]; } }
    for(int c=0;c<3;++c) acc1[(long)t*3+c]=a1[c];
  }
}

// ================= readout =================
__global__ void k_readout(int layer, const float* __restrict__ f0,
    const float* __restrict__ w_or_w1, const float* __restrict__ w2,
    const int* __restrict__ batch, float* __restrict__ out)
{
  int n = blockIdx.x*blockDim.x + threadIdx.x;
  if(n>=NN) return;
  float es=0.f;
  if(layer==0){
    for(int k=0;k<CH;++k) es += f0[(long)n*CH+k]*w_or_w1[k];
  } else {
    for(int j=0;j<16;++j){
      float h=0.f;
      for(int k=0;k<CH;++k) h += f0[(long)n*CH+k]*w_or_w1[k*16+j];
      es += siluf(h)*w2[j];
    }
  }
  atomicAdd(&out[batch[n]], es);
}

// ================= host driver =================
extern "C" void kernel_launch(void* const* d_in, const int* in_sizes, int n_in,
                              void* d_out, int out_size, void* d_ws, size_t ws_size,
                              hipStream_t stream)
{
  (void)in_sizes; (void)n_in; (void)out_size; (void)ws_size;
  // ---- input mapping (data arrays in setup_inputs order; params pytree sorted-key) ----
  const float* pos   = (const float*)d_in[0];
  const float* nattr = (const float*)d_in[1];
  const int*   eidx  = (const int*)  d_in[2];
  const int*   batch = (const int*)  d_in[3];
  const float* E0v   = (const float*)d_in[4];
  const float* Wemb  = (const float*)d_in[5];
  const float* L0_lin[4]  = {(const float*)d_in[6],(const float*)d_in[7],
                             (const float*)d_in[8],(const float*)d_in[9]};
  const float* L0_plin0   = (const float*)d_in[10];
  const float* L0_plin1   = (const float*)d_in[11];
  const float* L0_pw1_0   = (const float*)d_in[12];
  const float* L0_pw1_1   = (const float*)d_in[13];
  const float* L0_pw2_00  = (const float*)d_in[14];
  const float* L0_pw2_01  = (const float*)d_in[15];
  const float* L0_pw2_11  = (const float*)d_in[16];
  const float* L0_pw2_12  = (const float*)d_in[17];
  const float* L0_pw2_22  = (const float*)d_in[18];
  const float* L0_pw2_23  = (const float*)d_in[19];
  const float* L0_pw2_33  = (const float*)d_in[20];
  const float* L0_rad[4]  = {(const float*)d_in[21],(const float*)d_in[22],
                             (const float*)d_in[23],(const float*)d_in[24]};
  const float* L0_ro_w    = (const float*)d_in[25];
  const float* L0_skip    = (const float*)d_in[26];
  const float* L0_up0     = (const float*)d_in[27];
  const float* L1_lin[4]  = {(const float*)d_in[28],(const float*)d_in[29],
                             (const float*)d_in[30],(const float*)d_in[31]};
  const float* L1_plin0   = (const float*)d_in[32];
  const float* L1_pw1_0   = (const float*)d_in[33];
  const float* L1_pw2_00  = (const float*)d_in[34];
  const float* L1_pw2_11  = (const float*)d_in[35];
  const float* L1_pw2_22  = (const float*)d_in[36];
  const float* L1_pw2_33  = (const float*)d_in[37];
  const float* L1_rad[4]  = {(const float*)d_in[38],(const float*)d_in[39],
                             (const float*)d_in[40],(const float*)d_in[41]};
  const float* L1_ro_w1   = (const float*)d_in[42];
  const float* L1_ro_w2   = (const float*)d_in[43];
  const float* L1_skip    = (const float*)d_in[44];
  const float* L1_up0     = (const float*)d_in[45];
  const float* L1_up1     = (const float*)d_in[46];

  float* ws  = (float*)d_ws;
  float* CGt = ws + OFF_CG;
  float* Yb  = ws + OFF_Y;
  float* EFb = ws + OFF_EF;
  float* fA0 = ws + OFF_FA0;
  float* fB0 = ws + OFF_FB0;
  float* fB1 = ws + OFF_FB1;
  float* fC0 = ws + OFF_FC0;
  float* fu0 = ws + OFF_FU0;
  float* fu1 = ws + OFF_FU1;
  float* msg = ws + OFF_MSG;
  float* mid = ws + OFF_MID;
  float* ac0 = ws + OFF_AC0;
  float* ac1 = ws + OFF_AC1;
  float* sc  = ws + OFF_SC;
  float* out = (float*)d_out;
  const float invN = 1.0f/AVGN;

  k_zero<<<1,64,0,stream>>>(out, (long)NG);
  k_cg_init<<<1,64,0,stream>>>(CGt);
  k_geom<<<(NE+255)/256,256,0,stream>>>(pos, eidx, Yb, EFb);
  k_embed<<<(NN*CH+255)/256,256,0,stream>>>(nattr, Wemb, E0v, batch, fA0, out);

  // ================= layer 0 =================
  k_gemm_nodes<1,false><<<(NN+15)/16,128,0,stream>>>(fA0, L0_up0, fu0, nullptr, NN, 1.f);
  k_zero<<<512,256,0,stream>>>(msg, MSGSZ);
  k_edge<0><<<NE/16,128,0,stream>>>(CGt, Yb, EFb, fu0, nullptr,
      L0_rad[0], L0_rad[1], L0_rad[2], L0_rad[3], eidx, msg);
  k_gemm_nodes<1,false><<<(NN*1+15)/16,128,0,stream>>>(msg+MO0, L0_lin[0], mid+MO0, nullptr, NN*1, invN);
  k_gemm_nodes<3,false><<<(NN*3+15)/16,128,0,stream>>>(msg+MO1, L0_lin[1], mid+MO1, nullptr, NN*3, invN);
  k_gemm_nodes<5,false><<<(NN*5+15)/16,128,0,stream>>>(msg+MO2, L0_lin[2], mid+MO2, nullptr, NN*5, invN);
  k_gemm_nodes<7,false><<<(NN*7+15)/16,128,0,stream>>>(msg+MO3, L0_lin[3], mid+MO3, nullptr, NN*7, invN);
  k_node_products<<<(NN*CH+255)/256,256,0,stream>>>(0, nattr, fA0, L0_skip, mid,
      L0_pw1_0, L0_pw1_1, L0_pw2_00, L0_pw2_11, L0_pw2_22, L0_pw2_33,
      L0_pw2_01, L0_pw2_12, L0_pw2_23, CGt, ac0, ac1, sc);
  k_gemm_nodes<1,true><<<(NN+15)/16,128,0,stream>>>(ac0, L0_plin0, fB0, sc, NN, 1.f);
  k_gemm_nodes<3,false><<<(NN*3+15)/16,128,0,stream>>>(ac1, L0_plin1, fB1, nullptr, NN*3, 1.f);
  k_readout<<<(NN+255)/256,256,0,stream>>>(0, fB0, L0_ro_w, nullptr, batch, out);

  // ================= layer 1 =================
  k_gemm_nodes<1,false><<<(NN+15)/16,128,0,stream>>>(fB0, L1_up0, fu0, nullptr, NN, 1.f);
  k_gemm_nodes<3,false><<<(NN*3+15)/16,128,0,stream>>>(fB1, L1_up1, fu1, nullptr, NN*3, 1.f);
  k_zero<<<512,256,0,stream>>>(msg, MSGSZ);
  k_edge<1><<<NE/16,128,0,stream>>>(CGt, Yb, EFb, fu0, fu1,
      L1_rad[0], L1_rad[1], L1_rad[2], L1_rad[3], eidx, msg);
  k_gemm_nodes<1,false><<<(NN*1+15)/16,128,0,stream>>>(msg+MO0, L1_lin[0], mid+MO0, nullptr, NN*1, invN);
  k_gemm_nodes<3,false><<<(NN*3+15)/16,128,0,stream>>>(msg+MO1, L1_lin[1], mid+MO1, nullptr, NN*3, invN);
  k_gemm_nodes<5,false><<<(NN*5+15)/16,128,0,stream>>>(msg+MO2, L1_lin[2], mid+MO2, nullptr, NN*5, invN);
  k_gemm_nodes<7,false><<<(NN*7+15)/16,128,0,stream>>>(msg+MO3, L1_lin[3], mid+MO3, nullptr, NN*7, invN);
  k_node_products<<<(NN*CH+255)/256,256,0,stream>>>(1, nattr, fB0, L1_skip, mid,
      L1_pw1_0, nullptr, L1_pw2_00, L1_pw2_11, L1_pw2_22, L1_pw2_33,
      nullptr, nullptr, nullptr, CGt, ac0, nullptr, sc);
  k_gemm_nodes<1,true><<<(NN+15)/16,128,0,stream>>>(ac0, L1_plin0, fC0, sc, NN, 1.f);
  k_readout<<<(NN+255)/256,256,0,stream>>>(1, fC0, L1_ro_w1, L1_ro_w2, batch, out);
}
